// CNN_Encoder_20237885898939
// MI455X (gfx1250) — matrix operand order repaired and checked by reading, not compiled, not run on hardware
//
#include <hip/hip_runtime.h>
#include <hip/hip_bf16.h>
#include <math.h>

// ---------------------------------------------------------------------------
// Types for CDNA5 WMMA (gfx1250, wave32)
// ---------------------------------------------------------------------------
typedef __attribute__((ext_vector_type(16))) _Float16     v16h;
typedef __attribute__((ext_vector_type(8)))  float        v8f;
typedef __attribute__((ext_vector_type(4)))  unsigned int u32x4;

union HalfTile { v16h h; u32x4 q[2]; };

// Problem constants
static const int NB   = 4;     // batch
static const int HH   = 48;
static const int WW   = 48;
static const int HWSZ = 48 * 48;   // 2304
static const int CCH  = 64;        // channels
static const int HP   = 50;        // padded
static const int WP   = 50;
static const int NTIL = HWSZ / 16; // 144 row/col tiles

// ---------------------------------------------------------------------------
// 0) zero workspace region (padded f16 activation buffers -> zero borders)
// ---------------------------------------------------------------------------
__global__ void zero_kernel(unsigned int* p, int n) {
    int i = blockIdx.x * 256 + threadIdx.x;
    if (i < n) p[i] = 0u;
}

// ---------------------------------------------------------------------------
// 1) convert mid-conv weights f32 [6,64,64,3,3] -> f16 [k][dy][dx][o][c]
//    (A-operand friendly: for a fixed (k,dy,dx,o) the 64 c's are contiguous)
// ---------------------------------------------------------------------------
__global__ void wconv_kernel(const float* __restrict__ Wm, _Float16* __restrict__ wf) {
    int idx = blockIdx.x * 256 + threadIdx.x;          // 6*9*64*64 = 221184
    if (idx >= 6 * 9 * 64 * 64) return;
    int c = idx & 63;
    int o = (idx >> 6) & 63;
    int t = idx >> 12;                                  // t = k*9 + dy*3 + dx
    int k  = t / 9;
    int dd = t % 9;
    int dy = dd / 3, dx = dd % 3;
    float w = Wm[((((k * 64 + o) * 64) + c) * 3 + dy) * 3 + dx];
    wf[idx] = (_Float16)w;
}

// ---------------------------------------------------------------------------
// 2) Sobel edge on x3 [4,1,48,48] -> edge [4,2304]
// ---------------------------------------------------------------------------
__global__ void edge_kernel(const float* __restrict__ x3, float* __restrict__ edge) {
    int idx = blockIdx.x * 256 + threadIdx.x;           // 4*2304
    if (idx >= NB * HWSZ) return;
    int p = idx % HWSZ, b = idx / HWSZ;
    int y = p / WW, x = p % WW;
    const float* xb = x3 + b * HWSZ;
    auto at = [&](int yy, int xx) -> float {
        return (yy < 0 || yy >= HH || xx < 0 || xx >= WW) ? 0.f : xb[yy * WW + xx];
    };
    float ex = -at(y-1,x-1) + at(y-1,x+1) - 2.f*at(y,x-1) + 2.f*at(y,x+1) - at(y+1,x-1) + at(y+1,x+1);
    float ey = -at(y-1,x-1) - 2.f*at(y-1,x) - at(y-1,x+1) + at(y+1,x-1) + 2.f*at(y+1,x) + at(y+1,x+1);
    edge[idx] = sqrtf(ex * ex + ey * ey);
}

// ---------------------------------------------------------------------------
// 3) first-layer direct conv (variable Cin) + BN + LeakyReLU
//    writes the channel-last zero-padded f16 buffer [B][50][50][64]
// ---------------------------------------------------------------------------
__global__ void conv_first_kernel(const float* __restrict__ x,
                                  const float* __restrict__ Wc,
                                  const float* __restrict__ bc,
                                  const float* __restrict__ g,
                                  const float* __restrict__ be,
                                  const float* __restrict__ m,
                                  const float* __restrict__ var,
                                  _Float16* __restrict__ out16, int Cin) {
    int idx = blockIdx.x * 256 + threadIdx.x;           // 4*64*2304
    if (idx >= NB * CCH * HWSZ) return;
    int p = idx % HWSZ;
    int o = (idx / HWSZ) % CCH;
    int b = idx / (HWSZ * CCH);
    int y = p / WW, x0 = p % WW;
    float acc = 0.f;
    for (int c = 0; c < Cin; ++c) {
        const float* xb = x  + (b * Cin + c) * HWSZ;
        const float* wb = Wc + (o * Cin + c) * 9;
        #pragma unroll
        for (int dy = 0; dy < 3; ++dy) {
            int yy = y + dy - 1;
            if (yy < 0 || yy >= HH) continue;
            #pragma unroll
            for (int dx = 0; dx < 3; ++dx) {
                int xx = x0 + dx - 1;
                if (xx < 0 || xx >= WW) continue;
                acc += xb[yy * WW + xx] * wb[dy * 3 + dx];
            }
        }
    }
    float s = g[o] * rsqrtf(var[o] + 1e-5f);
    float val = (acc + bc[o]) * s + (be[o] - m[o] * s);
    val = fmaxf(val, 0.01f * val);                      // LeakyReLU(0.01)
    out16[(((b * HP) + y + 1) * WP + (x0 + 1)) * CCH + o] = (_Float16)val;
}

// ---------------------------------------------------------------------------
// 4) mid conv 64->64 as implicit GEMM with v_wmma_f32_16x16x32_f16.
//    grid = B*48*3 blocks, 128 threads (4 waves; wave w -> 16 output chans).
//    9 shifted 1x1 GEMMs x 2 K-chunks = 18 WMMAs per wave.
// ---------------------------------------------------------------------------
__global__ void conv_mid_kernel(const _Float16* __restrict__ in16,
                                const _Float16* __restrict__ wf,   // [9][64][64] this layer
                                const float* __restrict__ bias,
                                const float* __restrict__ g,
                                const float* __restrict__ be,
                                const float* __restrict__ m,
                                const float* __restrict__ var,
                                _Float16* __restrict__ out16,
                                float* __restrict__ outf /* nullable */) {
    int blk = blockIdx.x;
    int xt = blk % 3;
    int y  = (blk / 3) % HH;
    int b  = blk / (3 * HH);
    int wave = threadIdx.x >> 5;
    int lane = threadIdx.x & 31;
    int li = lane & 15, h = lane >> 4;
    int otile = wave;
    int x0 = xt * 16;

    v8f acc = {};
    #pragma unroll
    for (int dy = 0; dy < 3; ++dy) {
        #pragma unroll
        for (int dx = 0; dx < 3; ++dx) {
            const _Float16* wb = wf + (dy * 3 + dx) * 4096;  // [o][c]
            const _Float16* xb = in16 + (((b * HP) + (y + dy)) * WP + (x0 + dx + li)) * CCH;
            #pragma unroll
            for (int ch = 0; ch < 2; ++ch) {
                HalfTile A, Bm;
                int orow = otile * 16 + li;
                int cA = ch * 32 + 8 * h;                    // A: K pairs 2v(+8h), contiguous runs of 8
                A.q[0] = *(const u32x4*)(wb + orow * 64 + cA);
                A.q[1] = *(const u32x4*)(wb + orow * 64 + cA + 16);
                int cB = ch * 32 + 8 * h;                    // B: K pairs 2v(+8h), contiguous runs of 8
                Bm.q[0] = *(const u32x4*)(xb + cB);
                Bm.q[1] = *(const u32x4*)(xb + cB + 16);
                acc = __builtin_amdgcn_wmma_f32_16x16x32_f16(
                        false, A.h, false, Bm.h, (short)0, acc, false, false);
            }
        }
    }

    // epilogue: D[r] -> o = otile*16 + 8h + r (8 consecutive channels per lane)
    union { _Float16 hv[8]; u32x4 q; } pack;
    #pragma unroll
    for (int r = 0; r < 8; ++r) {
        int o = otile * 16 + 8 * h + r;
        float s = g[o] * rsqrtf(var[o] + 1e-5f);
        float val = (acc[r] + bias[o]) * s + (be[o] - m[o] * s);
        val = fmaxf(val, 0.01f * val);
        pack.hv[r] = (_Float16)val;
        if (outf) outf[(b * CCH + o) * HWSZ + y * WW + x0 + li] = val;
    }
    *(u32x4*)(out16 + (((b * HP) + (y + 1)) * WP + (x0 + li + 1)) * CCH
              + otile * 16 + 8 * h) = pack.q;
}

// ---------------------------------------------------------------------------
// 5) attention prep: per (branch,b,p): r = ||f||^2, v = Wv f + bv
// ---------------------------------------------------------------------------
__global__ void prep_attn_kernel(const float* __restrict__ d1,
                                 const float* __restrict__ d2,
                                 const float* __restrict__ d3,
                                 const float* __restrict__ Wv,
                                 const float* __restrict__ bv,
                                 float* __restrict__ rArr,
                                 float* __restrict__ vArr) {
    int idx = blockIdx.x * 256 + threadIdx.x;           // 3*4*2304
    if (idx >= 3 * NB * HWSZ) return;
    int p  = idx % HWSZ;
    int b  = (idx / HWSZ) % NB;
    int br = idx / (HWSZ * NB);
    const float* d = (br == 0 ? d1 : (br == 1 ? d2 : d3)) + (b * CCH) * HWSZ + p;
    float r = 0.f;
    float va[8];
    #pragma unroll
    for (int vc = 0; vc < 8; ++vc) va[vc] = bv[vc];
    for (int c = 0; c < CCH; ++c) {
        float f = d[c * HWSZ];
        r += f * f;
        #pragma unroll
        for (int vc = 0; vc < 8; ++vc) va[vc] += Wv[vc * 64 + c] * f;
    }
    rArr[idx] = r;
    #pragma unroll
    for (int vc = 0; vc < 8; ++vc) vArr[idx * 8 + vc] = va[vc];
}

// ---------------------------------------------------------------------------
// 6) fused flash-style attention + out-conv + multiply + edge add.
//    block = (b, I-rowtile); 4 waves split the 144 J-coltiles.
//    Gram per (I,J,branch): 2x v_wmma_f32_16x16x32_f16 over C=64.
// ---------------------------------------------------------------------------
__global__ void attn_kernel(const _Float16* __restrict__ f1,
                            const _Float16* __restrict__ f2,
                            const _Float16* __restrict__ f3,
                            const float* __restrict__ rArr,
                            const float* __restrict__ vArr,
                            const float* __restrict__ d1,
                            const float* __restrict__ d2,
                            const float* __restrict__ d3,
                            const float* __restrict__ edge,
                            const float* __restrict__ Wo,
                            const float* __restrict__ bo,
                            const float* __restrict__ gam,
                            float* __restrict__ out) {
    __shared__ float ldsS[4][3][16][16];   // per-wave similarity tiles
    __shared__ float ldsV[4][3][16][8];    // per-wave V tiles
    __shared__ float ldsAcc[3][16][8];     // block accumulators

    int blk = blockIdx.x;
    int I = blk % NTIL;
    int b = blk / NTIL;
    int wave = threadIdx.x >> 5;
    int lane = threadIdx.x & 31;
    int li = lane & 15, h = lane >> 4;

    for (int t = threadIdx.x; t < 3 * 16 * 8; t += 128) ((float*)ldsAcc)[t] = 0.f;
    __syncthreads();

    const _Float16* fptr[3] = { f1, f2, f3 };

    // gammas
    float g9[9];
    #pragma unroll
    for (int k = 0; k < 9; ++k) g9[k] = gam[k];
    float inv2s[3];
    #pragma unroll
    for (int k = 0; k < 3; ++k) inv2s[k] = 0.5f / (g9[k] * g9[k]);
    float cmat[3][3] = { { g9[6],       1.f + g9[7], 1.f + g9[8] },
                         { 1.f + g9[6], g9[7],       1.f + g9[8] },
                         { 1.f + g9[6], 1.f + g9[7], g9[8]       } };

    // A tiles: rows I*16+li, K=channel (f16, channel-last padded buffer)
    HalfTile A[3][2];
    int pi = I * 16 + li;
    int yi = pi / WW, xi = pi % WW;
    #pragma unroll
    for (int br = 0; br < 3; ++br) {
        const _Float16* base = fptr[br] + (((b * HP) + (yi + 1)) * WP + (xi + 1)) * CCH;
        #pragma unroll
        for (int ch = 0; ch < 2; ++ch) {
            int cA = ch * 32 + 8 * h;
            A[br][ch].q[0] = *(const u32x4*)(base + cA);
            A[br][ch].q[1] = *(const u32x4*)(base + cA + 16);
        }
    }
    // squared norms for this lane's D rows (i = I*16 + 8h + r)
    float ri[3][8];
    #pragma unroll
    for (int br = 0; br < 3; ++br)
        #pragma unroll
        for (int r = 0; r < 8; ++r)
            ri[br][r] = rArr[(br * NB + b) * HWSZ + I * 16 + 8 * h + r];

    float accv[3][4] = {};                 // lane owns (i=li, vc = 4h..4h+3)

    for (int J = wave; J < NTIL; J += 4) {
        int pj = J * 16 + li;
        int yj = pj / WW, xj = pj % WW;
        #pragma unroll
        for (int br = 0; br < 3; ++br) {
            const _Float16* base = fptr[br] + (((b * HP) + (yj + 1)) * WP + (xj + 1)) * CCH;
            HalfTile B0, B1;
            B0.q[0] = *(const u32x4*)(base + 8 * h);
            B0.q[1] = *(const u32x4*)(base + 8 * h + 16);
            B1.q[0] = *(const u32x4*)(base + 32 + 8 * h);
            B1.q[1] = *(const u32x4*)(base + 32 + 8 * h + 16);
            v8f gacc = {};
            gacc = __builtin_amdgcn_wmma_f32_16x16x32_f16(false, A[br][0].h, false, B0.h,
                                                          (short)0, gacc, false, false);
            gacc = __builtin_amdgcn_wmma_f32_16x16x32_f16(false, A[br][1].h, false, B1.h,
                                                          (short)0, gacc, false, false);
            float rj = rArr[(br * NB + b) * HWSZ + pj];   // col j = li
            #pragma unroll
            for (int r = 0; r < 8; ++r) {
                float d2 = fmaxf(ri[br][r] + rj - 2.f * gacc[r], 0.f);
                ldsS[wave][br][8 * h + r][li] = __expf(-d2 * inv2s[br]);
            }
        }
        // stage V tiles for this J (wave-private)
        for (int t = lane; t < 3 * 16 * 8; t += 32) {
            int vc = t & 7, j = (t >> 3) & 15, b2 = t >> 7;
            ldsV[wave][b2][j][vc] = vArr[(((b2 * NB + b) * HWSZ) + J * 16 + j) * 8 + vc];
        }
        // wave-internal LDS write->read ordering (CDNA5 split counter)
        asm volatile("s_wait_dscnt 0x0" ::: "memory");

        #pragma unroll
        for (int bo_ = 0; bo_ < 3; ++bo_) {
            #pragma unroll
            for (int j = 0; j < 16; ++j) {
                float pc = cmat[bo_][0] * ldsS[wave][0][li][j]
                         + cmat[bo_][1] * ldsS[wave][1][li][j]
                         + cmat[bo_][2] * ldsS[wave][2][li][j];
                const float* vv = &ldsV[wave][bo_][j][h * 4];
                accv[bo_][0] += pc * vv[0];
                accv[bo_][1] += pc * vv[1];
                accv[bo_][2] += pc * vv[2];
                accv[bo_][3] += pc * vv[3];
            }
        }
    }

    // combine across waves
    #pragma unroll
    for (int br = 0; br < 3; ++br)
        #pragma unroll
        for (int t = 0; t < 4; ++t)
            atomicAdd(&ldsAcc[br][li][h * 4 + t], accv[br][t]);
    __syncthreads();

    // epilogue: out-conv (8->64), multiply by d_i, add edge, store
    const float* dptr[3] = { d1, d2, d3 };
    for (int t = threadIdx.x; t < 3 * 64 * 16; t += 128) {
        int i  = t & 15;
        int o  = (t >> 4) & 63;
        int br = t >> 10;
        float gs = g9[3 + br];
        float val = bo[o];
        #pragma unroll
        for (int vc = 0; vc < 8; ++vc) val += Wo[o * 8 + vc] * (gs * ldsAcc[br][i][vc]);
        int p = I * 16 + i;
        float res = dptr[br][(b * CCH + o) * HWSZ + p] * val + edge[b * HWSZ + p];
        out[br * (NB * CCH * HWSZ) + (b * CCH + o) * HWSZ + p] = res;
    }
}

// ---------------------------------------------------------------------------
// launch
// ---------------------------------------------------------------------------
extern "C" void kernel_launch(void* const* d_in, const int* in_sizes, int n_in,
                              void* d_out, int out_size, void* d_ws, size_t ws_size,
                              hipStream_t stream) {
    const float* x1   = (const float*)d_in[0];
    const float* x2   = (const float*)d_in[1];
    const float* x3   = (const float*)d_in[2];
    const float* Wc1  = (const float*)d_in[3];
    const float* bc1  = (const float*)d_in[4];
    const float* Wc2  = (const float*)d_in[5];
    const float* bc2  = (const float*)d_in[6];
    const float* Wc3  = (const float*)d_in[7];
    const float* bc3  = (const float*)d_in[8];
    const float* bnfg = (const float*)d_in[9];
    const float* bnfb = (const float*)d_in[10];
    const float* bnfm = (const float*)d_in[11];
    const float* bnfv = (const float*)d_in[12];
    const float* Wm   = (const float*)d_in[13];
    const float* bm   = (const float*)d_in[14];
    const float* bnmg = (const float*)d_in[15];
    const float* bnmb = (const float*)d_in[16];
    const float* bnmm = (const float*)d_in[17];
    const float* bnmv = (const float*)d_in[18];
    const float* Wv   = (const float*)d_in[19];
    const float* bv   = (const float*)d_in[20];
    const float* Wo   = (const float*)d_in[21];
    const float* bo   = (const float*)d_in[22];
    const float* gam  = (const float*)d_in[23];

    // workspace carve (256B aligned)
    char* ws = (char*)d_ws;
    size_t off = 0;
    auto carve = [&](size_t bytes) -> char* {
        char* p = ws + off;
        off += (bytes + 255) & ~(size_t)255;
        return p;
    };
    const size_t ACT_BYTES = (size_t)NB * HP * WP * CCH * sizeof(_Float16); // 1.28MB
    char* actBase = carve(6 * ACT_BYTES);
    _Float16* actA[3];
    _Float16* actB[3];
    for (int br = 0; br < 3; ++br) {
        actA[br] = (_Float16*)(actBase + (size_t)br * ACT_BYTES);
        actB[br] = (_Float16*)(actBase + (size_t)(3 + br) * ACT_BYTES);
    }
    _Float16* wf = (_Float16*)carve((size_t)6 * 9 * 64 * 64 * sizeof(_Float16));
    float* dF[3];
    for (int br = 0; br < 3; ++br)
        dF[br] = (float*)carve((size_t)NB * CCH * HWSZ * sizeof(float));
    float* rA  = (float*)carve((size_t)3 * NB * HWSZ * sizeof(float));
    float* vA  = (float*)carve((size_t)3 * NB * HWSZ * 8 * sizeof(float));
    float* edg = (float*)carve((size_t)NB * HWSZ * sizeof(float));

    // 0) zero padded f16 activation buffers (borders must be 0)
    int nz = (int)(6 * ACT_BYTES / 4);
    zero_kernel<<<(nz + 255) / 256, 256, 0, stream>>>((unsigned int*)actBase, nz);
    // 1) weight conversion
    wconv_kernel<<<(6 * 9 * 64 * 64 + 255) / 256, 256, 0, stream>>>(Wm, wf);
    // 2) edge
    edge_kernel<<<(NB * HWSZ + 255) / 256, 256, 0, stream>>>(x3, edg);
    // 3) first-layer blocks
    int nf = NB * CCH * HWSZ;
    conv_first_kernel<<<(nf + 255) / 256, 256, 0, stream>>>(
        x1, Wc1, bc1, bnfg + 0, bnfb + 0, bnfm + 0, bnfv + 0, actA[0], 144);
    conv_first_kernel<<<(nf + 255) / 256, 256, 0, stream>>>(
        x2, Wc2, bc2, bnfg + 64, bnfb + 64, bnfm + 64, bnfv + 64, actA[1], 21);
    conv_first_kernel<<<(nf + 255) / 256, 256, 0, stream>>>(
        x3, Wc3, bc3, bnfg + 128, bnfb + 128, bnfm + 128, bnfv + 128, actA[2], 1);
    // 4) two WMMA mid conv blocks per branch
    int cg = NB * HH * 3;  // 576 blocks
    for (int br = 0; br < 3; ++br) {
        int k0 = 2 * br, k1 = 2 * br + 1;
        conv_mid_kernel<<<cg, 128, 0, stream>>>(
            actA[br], wf + (size_t)k0 * 9 * 4096, bm + k0 * 64,
            bnmg + k0 * 64, bnmb + k0 * 64, bnmm + k0 * 64, bnmv + k0 * 64,
            actB[br], (float*)nullptr);
        conv_mid_kernel<<<cg, 128, 0, stream>>>(
            actB[br], wf + (size_t)k1 * 9 * 4096, bm + k1 * 64,
            bnmg + k1 * 64, bnmb + k1 * 64, bnmm + k1 * 64, bnmv + k1 * 64,
            actA[br], dF[br]);
    }
    // 5) attention prep (norms + value proj)
    prep_attn_kernel<<<(3 * NB * HWSZ + 255) / 256, 256, 0, stream>>>(
        dF[0], dF[1], dF[2], Wv, bv, rA, vA);
    // 6) fused attention + out conv + edge, writes d_out
    attn_kernel<<<NB * NTIL, 128, 0, stream>>>(
        actA[0], actA[1], actA[2], rA, vA, dF[0], dF[1], dF[2],
        edg, Wo, bo, gam, (float*)d_out);
}
